// NeuralMemory_21698174779843
// MI455X (gfx1250) — matrix operand order repaired and checked by reading, not compiled, not run on hardware
//
#include <hip/hip_runtime.h>
#include <hip/hip_bf16.h>
#include <math.h>

// ---------------- problem constants ----------------
#define H_     64
#define EH_    128        // EF*H
#define T_     2048
#define B_     2
#define NT_    4096       // B*T tokens
#define MAXLR_ 0.1f

typedef __attribute__((ext_vector_type(16))) _Float16 v16h;
typedef __attribute__((ext_vector_type(8)))  float    v8f;
typedef __attribute__((ext_vector_type(4)))  int      v4i_;

// ---------------- CDNA5 async global->LDS path (guarded) ----------------
#if defined(__has_builtin)
#  if __has_builtin(__builtin_amdgcn_global_load_async_to_lds_b128)
#    define HAS_ASYNC_LDS 1
#  endif
#  if __has_builtin(__builtin_amdgcn_s_wait_asynccnt)
#    define HAS_WAIT_ASYNC 1
#  endif
#endif

typedef __attribute__((address_space(1))) v4i_ gv4i;  // global b128 payload
typedef __attribute__((address_space(3))) v4i_ lv4i;  // LDS b128 payload

__device__ __forceinline__ void cp_async16(void* lds, const void* gsrc) {
#ifdef HAS_ASYNC_LDS
  __builtin_amdgcn_global_load_async_to_lds_b128((gv4i*)gsrc, (lv4i*)lds, 0, 0);
#else
  *(float4*)lds = *(const float4*)gsrc;
#endif
}
__device__ __forceinline__ void wait_async() {
#ifdef HAS_ASYNC_LDS
#  ifdef HAS_WAIT_ASYNC
  __builtin_amdgcn_s_wait_asynccnt(0);
#  else
  asm volatile("s_wait_asynccnt 0" ::: "memory");
#  endif
#endif
}

// fast sigmoid: v_exp_f32 + v_add + v_rcp + (mul) — avoids the IEEE div sequence
__device__ __forceinline__ float sigmf(float x) {
  return __builtin_amdgcn_rcpf(1.0f + __expf(-x));
}

__device__ __forceinline__ v8f wmma_f16(v16h a, v16h b, v8f c) {
  // (neg_a, A, neg_b, B, c_mod, C, reuse_a, reuse_b) -> v_wmma_f32_16x16x32_f16
  return __builtin_amdgcn_wmma_f32_16x16x32_f16(false, a, false, b, (short)0, c,
                                                false, false);
}

// A fragment (16x32 f16, M x K): lane L holds row M=L&15; per CDNA5 ISA layout
// elems 0-7 -> K = e+8*hi, elems 8-15 -> K = e+8+8*hi  (hi = L>>4)
__device__ __forceinline__ int a_kmap(int e, int hi) {
  return e + 8 * hi + ((e >= 8) ? 8 : 0);
}

template <typename TSrc>
__device__ __forceinline__ v16h ldA(const TSrc* p, int sM, int sK, int lane) {
  int M = lane & 15, hi = lane >> 4;
  v16h a;
#pragma unroll
  for (int e = 0; e < 16; ++e) {
    int k = a_kmap(e, hi);
    a[e] = (_Float16)(float)p[M * sM + k * sK];
  }
  return a;
}

// A fragment scaled per-K by c[k] (folds the scan coefficient into the factor)
__device__ __forceinline__ v16h ldA_scaled(const _Float16* p, const float* c,
                                           int sM, int sK, int lane) {
  int M = lane & 15, hi = lane >> 4;
  v16h a;
#pragma unroll
  for (int e = 0; e < 16; ++e) {
    int k = a_kmap(e, hi);
    a[e] = (_Float16)((float)p[M * sM + k * sK] * c[k]);
  }
  return a;
}

// B fragment (32x16 f16, K x N): lane L holds column N=L&15; elem e -> K=a_kmap(e,hi)
template <typename TSrc>
__device__ __forceinline__ v16h ldB(const TSrc* p, int sK, int sN, int lane) {
  int N = lane & 15, hi = lane >> 4;
  v16h b;
#pragma unroll
  for (int e = 0; e < 16; ++e) {
    int k = a_kmap(e, hi);
    b[e] = (_Float16)(float)p[k * sK + N * sN];
  }
  return b;
}

// D layout: VGPR v, lane L -> D[v + 8*(L>>4)][L&15]

// ========== kernel 0: one-time f32->f16 conversion of all weights ==========
__global__ __launch_bounds__(256) void k_cvt(
    const float* __restrict__ Wf, const float* __restrict__ W1,
    const float* __restrict__ W2, _Float16* __restrict__ wfh,
    _Float16* __restrict__ w1h, _Float16* __restrict__ w2h) {
  int i = blockIdx.x * 256 + threadIdx.x;
  if (i < 195 * 64) wfh[i] = (_Float16)Wf[i];
  if (i < EH_ * H_) { w1h[i] = (_Float16)W1[i]; w2h[i] = (_Float16)W2[i]; }
}

// ================= kernel 1: fused projection + activations =================
__global__ __launch_bounds__(128) void k_proj(
    const float* __restrict__ x, const float* __restrict__ Wf,
    const _Float16* __restrict__ wfh,
    _Float16* __restrict__ qb, _Float16* __restrict__ kb,
    _Float16* __restrict__ vb, float* __restrict__ ab,
    float* __restrict__ thb, float* __restrict__ eb) {
  __shared__ __align__(16) float xs[16 * 64];
  __shared__ float ps[16 * 192];
  __shared__ float inrm[16 * 2];
  __shared__ float sc[16 * 3];
  const int tid = threadIdx.x, lane = tid & 31, wv = tid >> 5;
  const int t0 = blockIdx.x * 16;

  // async-stage the 16x64 f32 input tile into LDS (2 x b128 per thread)
  for (int i = tid * 4; i < 16 * 64; i += 128 * 4)
    cp_async16(xs + i, x + t0 * 64 + i);
  wait_async();
  __syncthreads();

  // scalar gate projections (W_fused rows 192..194)
  if (tid < 48) {
    int tok = tid / 3, wh = tid % 3;
    const float* wr = Wf + (192 + wh) * 64;
    float s = 0.f;
    for (int h = 0; h < 64; ++h) s += xs[tok * 64 + h] * wr[h];
    sc[tok * 3 + wh] = s;
  }

  // 12 tiles of 16 outputs (q,k,v): C(16x16) = X(16x64) @ Wf_chunk^T
  for (int i = 0; i < 3; ++i) {
    int n0 = (wv * 3 + i) * 16;
    v8f acc = {};
    for (int k0 = 0; k0 < 64; k0 += 32) {
      v16h a = ldA(xs + k0, 64, 1, lane);
      v16h b = ldB(wfh + n0 * 64 + k0, 1, 64, lane);  // B[k][n]=Wf[(n0+n)*64+k0+k]
      acc = wmma_f16(a, b, acc);
    }
    int N = lane & 15, hi = lane >> 4;
#pragma unroll
    for (int vv = 0; vv < 8; ++vv) {
      int M = vv + 8 * hi;
      float h = acc[vv];
      ps[M * 192 + n0 + N] = h * sigmf(h);  // silu for q,k,v alike
    }
  }
  __syncthreads();

  // l2 norms of silu(q), silu(k)
  if (tid < 32) {
    int tok = tid & 15, wh = tid >> 4;
    const float* pr = ps + tok * 192 + wh * 64;
    float ss = 0.f;
    for (int h = 0; h < 64; ++h) { float v = pr[h]; ss += v * v; }
    inrm[tok * 2 + wh] = __builtin_amdgcn_rcpf(fmaxf(sqrtf(ss), 1e-12f));
  }
  __syncthreads();

  for (int idx = tid; idx < 16 * 192; idx += 128) {
    int tok = idx / 192, col = idx % 192;
    float val = ps[idx];
    int g = (t0 + tok) * 64;
    if (col < 64)       qb[g + col]        = (_Float16)(val * inrm[tok * 2 + 0]);
    else if (col < 128) kb[g + col - 64]   = (_Float16)(val * inrm[tok * 2 + 1]);
    else                vb[g + col - 128]  = (_Float16)val;
  }
  if (tid < 48) {
    int tok = tid / 3, wh = tid % 3;
    float s = sigmf(sc[tok * 3 + wh]);
    int g = t0 + tok;
    if (wh == 0)      ab[g]  = s;
    else if (wh == 1) thb[g] = s * MAXLR_;
    else              eb[g]  = s;
  }
}

// ============ kernel 2: per-token rank-1 gradient factors ============
__global__ __launch_bounds__(128) void k_grad(
    const _Float16* __restrict__ kb, const _Float16* __restrict__ vb,
    const _Float16* __restrict__ w1h, const _Float16* __restrict__ w2h,
    _Float16* __restrict__ sb, _Float16* __restrict__ dhb,
    _Float16* __restrict__ rb) {
  __shared__ __align__(16) _Float16 ks[16 * 64];
  __shared__ _Float16 ss[16 * 128];
  __shared__ float    sp[16 * 128];
  __shared__ _Float16 rs[16 * 64];
  const int tid = threadIdx.x, lane = tid & 31, wv = tid >> 5;
  const int t0 = blockIdx.x * 16;

  // async-stage k tile (16x64 f16 = one b128 per thread)
  for (int i = tid * 8; i < 16 * 64; i += 128 * 8)
    cp_async16(ks + i, kb + t0 * 64 + i);
  wait_async();
  __syncthreads();

  // H = K(16x64) @ W1^T -> 16x128 ; s = silu(h), sp = silu'(h)
  for (int i = 0; i < 2; ++i) {
    int n0 = (wv * 2 + i) * 16;
    v8f acc = {};
    for (int k0 = 0; k0 < 64; k0 += 32) {
      v16h a = ldA(ks + k0, 64, 1, lane);
      v16h b = ldB(w1h + n0 * 64 + k0, 1, 64, lane);
      acc = wmma_f16(a, b, acc);
    }
    int N = lane & 15, hi = lane >> 4;
#pragma unroll
    for (int vv = 0; vv < 8; ++vv) {
      int M = vv + 8 * hi;
      float h = acc[vv], sg = sigmf(h), s = h * sg;
      ss[M * 128 + n0 + N] = (_Float16)s;
      sp[M * 128 + n0 + N] = sg * (1.0f + h * (1.0f - sg));
      sb[(t0 + M) * 128 + n0 + N] = (_Float16)s;
    }
  }
  __syncthreads();

  // Y = K + S @ W2^T ; r = (2/H)(y - v)
  {
    int n0 = wv * 16;
    v8f acc = {};
    for (int k0 = 0; k0 < 128; k0 += 32) {
      v16h a = ldA(ss + k0, 128, 1, lane);
      v16h b = ldB(w2h + n0 * 128 + k0, 1, 128, lane);
      acc = wmma_f16(a, b, acc);
    }
    int N = lane & 15, hi = lane >> 4;
#pragma unroll
    for (int vv = 0; vv < 8; ++vv) {
      int M = vv + 8 * hi;
      float y = acc[vv] + (float)ks[M * 64 + n0 + N];
      float r = (2.0f / 64.0f) * (y - (float)vb[(t0 + M) * 64 + n0 + N]);
      rs[M * 64 + n0 + N] = (_Float16)r;
      rb[(t0 + M) * 64 + n0 + N] = (_Float16)r;
    }
  }
  __syncthreads();

  // DS = R(16x64) @ W2(64x128) ; dh = ds * silu'(h)
  for (int i = 0; i < 2; ++i) {
    int n0 = (wv * 2 + i) * 16;
    v8f acc = {};
    for (int k0 = 0; k0 < 64; k0 += 32) {
      v16h a = ldA(rs + k0, 64, 1, lane);
      v16h b = ldB(w2h + k0 * 128 + n0, 128, 1, lane);
      acc = wmma_f16(a, b, acc);
    }
    int N = lane & 15, hi = lane >> 4;
#pragma unroll
    for (int vv = 0; vv < 8; ++vv) {
      int M = vv + 8 * hi;
      float dh = acc[vv] * sp[M * 128 + n0 + N];
      dhb[(t0 + M) * 128 + n0 + N] = (_Float16)dh;
    }
  }
}

// ========= kernel 3: backward scalar scan for coefficients c_t, D_T =========
// W_T = D_T*W0 - sum_t c_t g_t, c_t = theta_t*F_t, F_t = Dsuf_t + eta_{t+1}F_{t+1}
__global__ __launch_bounds__(32) void k_coeff(
    const float* __restrict__ ab, const float* __restrict__ thb,
    const float* __restrict__ eb, float* __restrict__ cb,
    float* __restrict__ Dbar) {
  __shared__ float Dper[B_];
  int b = threadIdx.x;
  if (b < B_) {
    const int base = b * T_;
    float F = 1.0f, Dsuf = 1.0f;
    cb[base + T_ - 1] = thb[base + T_ - 1] * F * (1.0f / B_);
    for (int t = T_ - 2; t >= 0; --t) {
      Dsuf *= (1.0f - ab[base + t + 1]);
      F = Dsuf + eb[base + t + 1] * F;
      cb[base + t] = thb[base + t] * F * (1.0f / B_);
    }
    Dper[b] = (1.0f - ab[base]) * Dsuf;
  }
  __syncthreads();
  if (b == 0) Dbar[0] = 0.5f * (Dper[0] + Dper[1]);
}

// ======= kernel 4: W1n/W2n via token-dimension GEMMs (K = 4096 tokens) =======
__global__ __launch_bounds__(32) void k_update(
    const _Float16* __restrict__ kb, const _Float16* __restrict__ sb,
    const _Float16* __restrict__ dhb, const _Float16* __restrict__ rb,
    const float* __restrict__ cb, const float* __restrict__ Dbar_p,
    const float* __restrict__ W1, const float* __restrict__ W2,
    _Float16* __restrict__ W1nh, _Float16* __restrict__ W2nh) {
  const int lane = threadIdx.x;
  const int bi = blockIdx.x;
  const float Dbar = Dbar_p[0];
  v8f acc = {};
  if (bi < 32) {  // W1n (128x64): Dbar*W1 - sum_t (c_t dh_t) k_t^T
    int m0 = (bi >> 2) * 16, n0 = (bi & 3) * 16;
    for (int t0 = 0; t0 < NT_; t0 += 32) {
      if (t0 + 64 < NT_) {
        __builtin_prefetch(dhb + (t0 + 64) * 128 + m0, 0, 0);
        __builtin_prefetch(kb + (t0 + 64) * 64 + n0, 0, 0);
      }
      v16h a = ldA_scaled(dhb + t0 * 128 + m0, cb + t0, 1, 128, lane);
      v16h b = ldB(kb + t0 * 64 + n0, 64, 1, lane);
      acc = wmma_f16(a, b, acc);
    }
    int N = lane & 15, hi = lane >> 4;
#pragma unroll
    for (int vv = 0; vv < 8; ++vv) {
      int M = vv + 8 * hi;
      W1nh[(m0 + M) * 64 + n0 + N] =
          (_Float16)(Dbar * W1[(m0 + M) * 64 + n0 + N] - acc[vv]);
    }
  } else {        // W2n (64x128): Dbar*W2 - sum_t (c_t r_t) s_t^T
    int idx = bi - 32;
    int m0 = (idx >> 3) * 16, n0 = (idx & 7) * 16;
    for (int t0 = 0; t0 < NT_; t0 += 32) {
      if (t0 + 64 < NT_) {
        __builtin_prefetch(rb + (t0 + 64) * 64 + m0, 0, 0);
        __builtin_prefetch(sb + (t0 + 64) * 128 + n0, 0, 0);
      }
      v16h a = ldA_scaled(rb + t0 * 64 + m0, cb + t0, 1, 64, lane);
      v16h b = ldB(sb + t0 * 128 + n0, 128, 1, lane);
      acc = wmma_f16(a, b, acc);
    }
    int N = lane & 15, hi = lane >> 4;
#pragma unroll
    for (int vv = 0; vv < 8; ++vv) {
      int M = vv + 8 * hi;
      W2nh[(m0 + M) * 128 + n0 + N] =
          (_Float16)(Dbar * W2[(m0 + M) * 128 + n0 + N] - acc[vv]);
    }
  }
}

// ============== kernel 5: retrieval out = q + W2n silu(W1n q) ==============
__global__ __launch_bounds__(128) void k_retr(
    const _Float16* __restrict__ qb, const _Float16* __restrict__ W1nh,
    const _Float16* __restrict__ W2nh, float* __restrict__ out) {
  __shared__ __align__(16) _Float16 qs[16 * 64];
  __shared__ _Float16 sr[16 * 128];
  const int tid = threadIdx.x, lane = tid & 31, wv = tid >> 5;
  const int t0 = blockIdx.x * 16;

  for (int i = tid * 8; i < 16 * 64; i += 128 * 8)
    cp_async16(qs + i, qb + t0 * 64 + i);
  wait_async();
  __syncthreads();

  for (int i = 0; i < 2; ++i) {
    int n0 = (wv * 2 + i) * 16;
    v8f acc = {};
    for (int k0 = 0; k0 < 64; k0 += 32) {
      v16h a = ldA(qs + k0, 64, 1, lane);
      v16h b = ldB(W1nh + n0 * 64 + k0, 1, 64, lane);
      acc = wmma_f16(a, b, acc);
    }
    int N = lane & 15, hi = lane >> 4;
#pragma unroll
    for (int vv = 0; vv < 8; ++vv) {
      int M = vv + 8 * hi;
      float h = acc[vv];
      sr[M * 128 + n0 + N] = (_Float16)(h * sigmf(h));
    }
  }
  __syncthreads();
  {
    int n0 = wv * 16;
    v8f acc = {};
    for (int k0 = 0; k0 < 128; k0 += 32) {
      v16h a = ldA(sr + k0, 128, 1, lane);
      v16h b = ldB(W2nh + n0 * 128 + k0, 1, 128, lane);
      acc = wmma_f16(a, b, acc);
    }
    int N = lane & 15, hi = lane >> 4;
#pragma unroll
    for (int vv = 0; vv < 8; ++vv) {
      int M = vv + 8 * hi;
      out[(t0 + M) * 64 + n0 + N] = (float)qs[M * 64 + n0 + N] + acc[vv];
    }
  }
}

// ============================== launcher ==============================
extern "C" void kernel_launch(void* const* d_in, const int* in_sizes, int n_in,
                              void* d_out, int out_size, void* d_ws,
                              size_t ws_size, hipStream_t stream) {
  (void)in_sizes; (void)n_in; (void)out_size; (void)ws_size;
  const float* x  = (const float*)d_in[0];
  const float* Wf = (const float*)d_in[1];
  const float* W1 = (const float*)d_in[2];
  const float* W2 = (const float*)d_in[3];
  float* out = (float*)d_out;

  char* ws = (char*)d_ws;
  size_t off = 0;
  auto alloc = [&](size_t bytes) -> void* {
    void* p = (void*)(ws + off);
    off += (bytes + 255) & ~(size_t)255;
    return p;
  };
  _Float16* qb   = (_Float16*)alloc((size_t)NT_ * 64 * sizeof(_Float16));
  _Float16* kb   = (_Float16*)alloc((size_t)NT_ * 64 * sizeof(_Float16));
  _Float16* vb   = (_Float16*)alloc((size_t)NT_ * 64 * sizeof(_Float16));
  _Float16* sb   = (_Float16*)alloc((size_t)NT_ * 128 * sizeof(_Float16));
  _Float16* dhb  = (_Float16*)alloc((size_t)NT_ * 128 * sizeof(_Float16));
  _Float16* rb   = (_Float16*)alloc((size_t)NT_ * 64 * sizeof(_Float16));
  float* ab   = (float*)alloc((size_t)NT_ * sizeof(float));
  float* thb  = (float*)alloc((size_t)NT_ * sizeof(float));
  float* eb   = (float*)alloc((size_t)NT_ * sizeof(float));
  float* cb   = (float*)alloc((size_t)NT_ * sizeof(float));
  _Float16* wfh  = (_Float16*)alloc((size_t)195 * 64 * sizeof(_Float16));
  _Float16* w1h  = (_Float16*)alloc((size_t)EH_ * H_ * sizeof(_Float16));
  _Float16* w2h  = (_Float16*)alloc((size_t)H_ * EH_ * sizeof(_Float16));
  _Float16* w1nh = (_Float16*)alloc((size_t)EH_ * H_ * sizeof(_Float16));
  _Float16* w2nh = (_Float16*)alloc((size_t)H_ * EH_ * sizeof(_Float16));
  float* Dbar = (float*)alloc(256);

  const int ntiles = NT_ / 16;  // 256
  k_cvt<<<(195 * 64 + 255) / 256, 256, 0, stream>>>(Wf, W1, W2, wfh, w1h, w2h);
  k_proj<<<ntiles, 128, 0, stream>>>(x, Wf, wfh, qb, kb, vb, ab, thb, eb);
  k_grad<<<ntiles, 128, 0, stream>>>(kb, vb, w1h, w2h, sb, dhb, rb);
  k_coeff<<<1, 32, 0, stream>>>(ab, thb, eb, cb, Dbar);
  k_update<<<64, 32, 0, stream>>>(kb, sb, dhb, rb, cb, Dbar, W1, W2, w1nh, w2nh);
  k_retr<<<ntiles, 128, 0, stream>>>(qb, w1nh, w2nh, out);
}
